// AllInOne_lora_88424786690153
// MI455X (gfx1250) — matrix operand order repaired and checked by reading, not compiled, not run on hardware
//
#include <hip/hip_runtime.h>
#include <hip/hip_bf16.h>
#include <math.h>

// ---------------------------------------------------------------------------
// MoE patch-conv + shared expert + attention-weighted concat-conv, gfx1250.
// GEMM stages: v_wmma_f32_16x16x32_bf16 with 2x2 register tiling per wave
// (4 WMMAs per 2 ds + 2 global 32B fragment loads). Weights pre-packed to
// bf16 fragment order; im2col A tiles staged in LDS, shared by 4 waves.
// ---------------------------------------------------------------------------

#define NB   32            // batch
#define NC1  64            // in channels
#define NH   128           // H = W
#define NC2  128           // c_inter
#define NE   8             // experts
#define NK   2             // top-k
#define HM   32            // expert-conv out spatial (128/4), also H/P2
#define NM   (HM*HM)       // 1024 spatial positions per image
#define KEXP (NC1*7*7)     // 3136 -> 98 k-steps of 32
#define KSH  (NC1*4*4)     // 1024 -> 32 k-steps
#define KCC  (3*NC2*9)     // 3456 -> 108 k-steps
#define KTI_EXP 98
#define KTI_SH  32
#define KTI_CC  108
#define CHUNK_EXP 14       // 7 chunks;  LDS = 14*512*4 = 28 KB
#define CHUNK_SH  16       // 2 chunks;  LDS = 16*512*4 = 32 KB
#define CHUNK_CC  18       // 6 chunks;  LDS = 18*512*4 = 36 KB
#define EPS_LN 1e-6f

typedef __bf16 v16bf __attribute__((ext_vector_type(16)));
typedef float  v8f   __attribute__((ext_vector_type(8)));

// A-fragment K index (16-bit 16x32 A, CDNA5 ISA §7.12.2)
__device__ __forceinline__ int kA(int lane, int h) {
  return ((h & 8) << 1) + ((lane & 16) ? 8 : 0) + (h & 7);
}
// B-fragment K index (same table order as A)
__device__ __forceinline__ int kB(int lane, int h) {
  return ((h & 8) << 1) + ((lane & 16) ? 8 : 0) + (h & 7);
}
__device__ __forceinline__ unsigned pack2bf(float a, float b) {
  union { __bf16 h[2]; unsigned u; } p;
  p.h[0] = (__bf16)a; p.h[1] = (__bf16)b;
  return p.u;
}
#define WMMA_BF16(A, Bm, C) \
  __builtin_amdgcn_wmma_f32_16x16x32_bf16(false, (A), false, (Bm), (short)0, (C), false, false)

// ---------------------------------------------------------------------------
// Weight packers: fragment-ordered bf16, layout [...][ntile][kti][lane][16].
// ---------------------------------------------------------------------------
__global__ void moe_pack_wc(const float* __restrict__ Wc, __bf16* __restrict__ p) {
  const int idx = blockIdx.x * 256 + threadIdx.x;        // E*8*98*512
  if (idx >= NE * 8 * KTI_EXP * 512) return;
  const int h    = idx & 15;
  const int lane = (idx >> 4) & 31;
  const int kti  = (idx >> 9) % KTI_EXP;
  const int nt   = ((idx >> 9) / KTI_EXP) & 7;
  const int e    = idx / (8 * KTI_EXP * 512);
  const int k = kti * 32 + kB(lane, h);
  const int n = nt * 16 + (lane & 15);
  const int c = k / 49, r = k - c * 49;
  p[idx] = (__bf16)Wc[(((size_t)e * NC2 + n) * NC1 + c) * 49 + r];
}

__global__ void moe_pack_wsh(const float* __restrict__ Wsh, __bf16* __restrict__ p) {
  const int idx = blockIdx.x * 256 + threadIdx.x;        // 8*32*512
  if (idx >= 8 * KTI_SH * 512) return;
  const int h    = idx & 15;
  const int lane = (idx >> 4) & 31;
  const int kti  = (idx >> 9) % KTI_SH;
  const int nt   = (idx >> 9) / KTI_SH;
  const int k = kti * 32 + kB(lane, h);
  const int n = nt * 16 + (lane & 15);
  p[idx] = (__bf16)Wsh[(size_t)k * NC2 + n];
}

__global__ void moe_pack_wcc(const float* __restrict__ Wcc, __bf16* __restrict__ p) {
  const int idx = blockIdx.x * 256 + threadIdx.x;        // 8*108*512
  if (idx >= 8 * KTI_CC * 512) return;
  const int h    = idx & 15;
  const int lane = (idx >> 4) & 31;
  const int kti  = (idx >> 9) % KTI_CC;
  const int nt   = (idx >> 9) / KTI_CC;
  const int k = kti * 32 + kB(lane, h);
  const int n = nt * 16 + (lane & 15);
  p[idx] = (__bf16)Wcc[(size_t)n * KCC + k];
}

// ---------------------------------------------------------------------------
// K1: gating logits.  logits[b][e] = sum_p mean_c(x[b,c,p]) * w_gate[p][e]
// ---------------------------------------------------------------------------
__global__ void moe_gating_logits(const float* __restrict__ x,
                                  const float* __restrict__ wg,
                                  float* __restrict__ logits) {
  const int b = blockIdx.x, tid = threadIdx.x;
  const float* xb = x + (size_t)b * NC1 * NH * NH;
  float acc[NE];
#pragma unroll
  for (int e = 0; e < NE; ++e) acc[e] = 0.0f;
  for (int p = tid; p < NH * NH; p += 256) {
    float s = 0.0f;
#pragma unroll 8
    for (int c = 0; c < NC1; ++c) s += xb[c * (NH * NH) + p];
    s *= (1.0f / NC1);
    const float* w = wg + (size_t)p * NE;
#pragma unroll
    for (int e = 0; e < NE; ++e) acc[e] += s * w[e];
  }
  __shared__ float red[256 * NE];
#pragma unroll
  for (int e = 0; e < NE; ++e) red[tid * NE + e] = acc[e];
  __syncthreads();
  for (int s = 128; s > 0; s >>= 1) {
    if (tid < s)
#pragma unroll
      for (int e = 0; e < NE; ++e) red[tid * NE + e] += red[(tid + s) * NE + e];
    __syncthreads();
  }
  if (tid < NE) logits[b * NE + tid] = red[tid];
}

// ---------------------------------------------------------------------------
// K2: top-2 + softmax over the two selected logits.
// ---------------------------------------------------------------------------
__global__ void moe_top2(const float* __restrict__ logits,
                         int* __restrict__ sel, float* __restrict__ gates) {
  const int b = threadIdx.x;
  if (b >= NB) return;
  float l[NE];
#pragma unroll
  for (int e = 0; e < NE; ++e) l[e] = logits[b * NE + e];
  int i0 = 0;
#pragma unroll
  for (int e = 1; e < NE; ++e) if (l[e] > l[i0]) i0 = e;   // ties -> lowest idx
  int i1 = (i0 == 0) ? 1 : 0;
#pragma unroll
  for (int e = 0; e < NE; ++e) if (e != i0 && l[e] > l[i1]) i1 = e;
  const float d  = expf(l[i1] - l[i0]);
  const float g0 = 1.0f / (1.0f + d);
  sel[b * NK + 0] = i0;  sel[b * NK + 1] = i1;
  gates[b * NK + 0] = g0;  gates[b * NK + 1] = 1.0f - g0;
}

// ---------------------------------------------------------------------------
// K3: expert 7x7/s4/p3 conv, implicit GEMM M=1024 N=128 K=3136.
// Block = 128 thr (4 waves), block tile M=32 x N=128. Wave w owns M-tiles
// {0,1} x N-tiles {2w,2w+1}: 4 accumulators, 4 WMMAs per k-step.
// grid = (M/32 = 32, bk = 64).
// ---------------------------------------------------------------------------
__global__ void moe_expert_conv_wmma(const float* __restrict__ x,
                                     const __bf16* __restrict__ pwc,
                                     const float* __restrict__ bc,
                                     const int*   __restrict__ sel,
                                     float* __restrict__ y) {
  __shared__ unsigned ldsU[CHUNK_EXP * 512];   // [ktl][mtile][lane][h/2]
  const int tid = threadIdx.x;
  const int lane = tid & 31, w = tid >> 5;
  const int m0 = blockIdx.x * 32;
  const int bk = blockIdx.y;
  const int b  = bk >> 1;
  const int e  = sel[bk];
  const float*  xb  = x + (size_t)b * NC1 * NH * NH;
  const __bf16* pw0 = pwc + ((size_t)e * 8 + 2 * w + 0) * KTI_EXP * 512;
  const __bf16* pw1 = pwc + ((size_t)e * 8 + 2 * w + 1) * KTI_EXP * 512;

  v8f acc00 = {}, acc01 = {}, acc10 = {}, acc11 = {};
  for (int kt0 = 0; kt0 < KTI_EXP; kt0 += CHUNK_EXP) {
    __builtin_prefetch(&pw0[(size_t)kt0 * 512 + lane * 16], 0, 0);
    __builtin_prefetch(&pw1[(size_t)kt0 * 512 + lane * 16], 0, 0);
    // phase 1: build 32-row A chunk in LDS (bf16 pairs)
    for (int j = tid; j < CHUNK_EXP * 512; j += 128) {
      const int ktl = j >> 9;
      const int mi  = (j >> 8) & 1;
      const int rem = j & 255;
      const int li  = rem >> 3;
      const int h   = (rem & 7) << 1;
      const int m = m0 + mi * 16 + (li & 15);
      const int oy = m >> 5, ox = m & 31;
      const int k0 = (kt0 + ktl) * 32 + kA(li, h);
      float v[2];
#pragma unroll
      for (int q = 0; q < 2; ++q) {
        const int k  = k0 + q;
        const int c  = k / 49;
        const int r  = k - c * 49;
        const int di = r / 7, dj = r - (r / 7) * 7;
        const int iy = oy * 4 - 3 + di;
        const int ix = ox * 4 - 3 + dj;
        v[q] = (iy >= 0 && iy < NH && ix >= 0 && ix < NH)
                   ? xb[(c << 14) + (iy << 7) + ix] : 0.0f;
      }
      ldsU[j] = pack2bf(v[0], v[1]);
    }
    __syncthreads();
    // phase 2: 4 waves, 4 WMMAs per k-step
    const int klim = min(CHUNK_EXP, KTI_EXP - kt0);
    for (int c = 0; c < klim; ++c) {
      const v16bf a0 = *(const v16bf*)&ldsU[((c * 2 + 0) * 32 + lane) * 8];
      const v16bf a1 = *(const v16bf*)&ldsU[((c * 2 + 1) * 32 + lane) * 8];
      const v16bf b0 = *(const v16bf*)&pw0[((size_t)(kt0 + c) * 32 + lane) * 16];
      const v16bf b1 = *(const v16bf*)&pw1[((size_t)(kt0 + c) * 32 + lane) * 16];
      acc00 = WMMA_BF16(a0, b0, acc00);
      acc01 = WMMA_BF16(a0, b1, acc01);
      acc10 = WMMA_BF16(a1, b0, acc10);
      acc11 = WMMA_BF16(a1, b1, acc11);
    }
    __syncthreads();
  }

  const int mlo = m0 + ((lane >> 4) << 3);
  const int nlo = (lane & 15);
  v8f accs[4] = {acc00, acc01, acc10, acc11};
#pragma unroll
  for (int t = 0; t < 4; ++t) {
    const int mi = t >> 1, nj = t & 1;
    const int n = (2 * w + nj) * 16 + nlo;
    const int mbase = mlo + mi * 16;
    const float bias = bc[e * NC2 + n];
    float* yo = y + ((size_t)bk * NC2 + n) * NM;
#pragma unroll
    for (int r = 0; r < 8; ++r) yo[mbase + r] = accs[t][r] + bias;
  }
}

// ---------------------------------------------------------------------------
// K4: channel LayerNorm over C2 + gate-weighted combine. Thread per (b,m).
// ---------------------------------------------------------------------------
__global__ void moe_ln_combine(const float* __restrict__ y,
                               const int*   __restrict__ sel,
                               const float* __restrict__ gates,
                               const float* __restrict__ gamma,
                               const float* __restrict__ beta,
                               float* __restrict__ moe) {
  const int idx = blockIdx.x * blockDim.x + threadIdx.x;
  if (idx >= NB * NM) return;
  const int b = idx >> 10, m = idx & (NM - 1);
  const int   e0 = sel[b * NK + 0], e1 = sel[b * NK + 1];
  const float g0 = gates[b * NK + 0], g1 = gates[b * NK + 1];
  const float* y0 = y + ((size_t)(b * NK + 0) * NC2) * NM + m;
  const float* y1 = y + ((size_t)(b * NK + 1) * NC2) * NM + m;

  float s0 = 0.f, q0 = 0.f, s1 = 0.f, q1 = 0.f;
#pragma unroll 4
  for (int c = 0; c < NC2; ++c) {
    const float v0 = y0[c * NM], v1 = y1[c * NM];
    s0 += v0; q0 += v0 * v0;
    s1 += v1; q1 += v1 * v1;
  }
  const float u0 = s0 * (1.0f / NC2), u1 = s1 * (1.0f / NC2);
  const float i0 = rsqrtf(q0 * (1.0f / NC2) - u0 * u0 + EPS_LN);
  const float i1 = rsqrtf(q1 * (1.0f / NC2) - u1 * u1 + EPS_LN);

  float* mo = moe + (size_t)b * NC2 * NM + m;
#pragma unroll 4
  for (int c = 0; c < NC2; ++c) {
    const float n0v = (y0[c * NM] - u0) * i0 * gamma[e0 * NC2 + c] + beta[e0 * NC2 + c];
    const float n1v = (y1[c * NM] - u1) * i1 * gamma[e1 * NC2 + c] + beta[e1 * NC2 + c];
    mo[c * NM] = g0 * n0v + g1 * n1v;
  }
}

// ---------------------------------------------------------------------------
// K5: shared expert — 4x4 patchify GEMM [1024 x 1024] @ [1024 x 128] + GELU.
// Same 2x2 wave tiling. grid = (32, B), 128 threads.
// ---------------------------------------------------------------------------
__global__ void moe_shared_expert_wmma(const float* __restrict__ x,
                                       const __bf16* __restrict__ pwsh,
                                       const float* __restrict__ bsh,
                                       float* __restrict__ sh) {
  __shared__ unsigned ldsU[CHUNK_SH * 512];
  const int tid = threadIdx.x;
  const int lane = tid & 31, w = tid >> 5;
  const int m0 = blockIdx.x * 32;
  const int b  = blockIdx.y;
  const float*  xb  = x + (size_t)b * NC1 * NH * NH;
  const __bf16* pw0 = pwsh + (size_t)(2 * w + 0) * KTI_SH * 512;
  const __bf16* pw1 = pwsh + (size_t)(2 * w + 1) * KTI_SH * 512;

  v8f acc00 = {}, acc01 = {}, acc10 = {}, acc11 = {};
  for (int kt0 = 0; kt0 < KTI_SH; kt0 += CHUNK_SH) {
    __builtin_prefetch(&pw0[(size_t)kt0 * 512 + lane * 16], 0, 0);
    __builtin_prefetch(&pw1[(size_t)kt0 * 512 + lane * 16], 0, 0);
    for (int j = tid; j < CHUNK_SH * 512; j += 128) {
      const int ktl = j >> 9;
      const int mi  = (j >> 8) & 1;
      const int rem = j & 255;
      const int li  = rem >> 3;
      const int h   = (rem & 7) << 1;
      const int t = m0 + mi * 16 + (li & 15);
      const int ph = t >> 5, pcol = t & 31;
      const int k0 = (kt0 + ktl) * 32 + kA(li, h);
      float v[2];
#pragma unroll
      for (int q = 0; q < 2; ++q) {
        const int k = k0 + q;                  // feature = c*16 + i*4 + j
        const int c = k >> 4, i = (k >> 2) & 3, jj = k & 3;
        v[q] = xb[(c << 14) + (((ph << 2) + i) << 7) + (pcol << 2) + jj];
      }
      ldsU[j] = pack2bf(v[0], v[1]);
    }
    __syncthreads();
    for (int c = 0; c < CHUNK_SH; ++c) {
      const v16bf a0 = *(const v16bf*)&ldsU[((c * 2 + 0) * 32 + lane) * 8];
      const v16bf a1 = *(const v16bf*)&ldsU[((c * 2 + 1) * 32 + lane) * 8];
      const v16bf b0 = *(const v16bf*)&pw0[((size_t)(kt0 + c) * 32 + lane) * 16];
      const v16bf b1 = *(const v16bf*)&pw1[((size_t)(kt0 + c) * 32 + lane) * 16];
      acc00 = WMMA_BF16(a0, b0, acc00);
      acc01 = WMMA_BF16(a0, b1, acc01);
      acc10 = WMMA_BF16(a1, b0, acc10);
      acc11 = WMMA_BF16(a1, b1, acc11);
    }
    __syncthreads();
  }

  const int mlo = m0 + ((lane >> 4) << 3);
  const int nlo = (lane & 15);
  v8f accs[4] = {acc00, acc01, acc10, acc11};
#pragma unroll
  for (int t = 0; t < 4; ++t) {
    const int mi = t >> 1, nj = t & 1;
    const int n = (2 * w + nj) * 16 + nlo;
    const int mbase = mlo + mi * 16;
    const float bias = bsh[n];
    float* so = sh + ((size_t)b * NC2 + n) * NM;
#pragma unroll
    for (int r = 0; r < 8; ++r) {
      const float z = accs[t][r] + bias;
      const float g = 0.5f * z * (1.0f + tanhf(0.7978845608028654f *
                                               (z + 0.044715f * z * z * z)));
      so[mbase + r] = g;
    }
  }
}

// ---------------------------------------------------------------------------
// K6: per-batch dots  s1[b]=sum(moe*moe), s2[b]=sum(moe*sh)
// ---------------------------------------------------------------------------
__global__ void moe_batch_dots(const float* __restrict__ moe,
                               const float* __restrict__ sh,
                               float* __restrict__ s1s2) {
  const int b = blockIdx.x, tid = threadIdx.x;
  const float* mp = moe + (size_t)b * NC2 * NM;
  const float* sp = sh  + (size_t)b * NC2 * NM;
  float a = 0.f, c = 0.f;
  for (int i = tid; i < NC2 * NM; i += 256) {
    const float mv = mp[i], sv = sp[i];
    a += mv * mv;  c += mv * sv;
  }
  __shared__ float l1[256], l2[256];
  l1[tid] = a; l2[tid] = c;
  __syncthreads();
  for (int s = 128; s > 0; s >>= 1) {
    if (tid < s) { l1[tid] += l1[tid + s]; l2[tid] += l2[tid + s]; }
    __syncthreads();
  }
  if (tid == 0) { s1s2[b] = l1[0]; s1s2[NB + b] = l2[0]; }
}

// ---------------------------------------------------------------------------
// K7: softmax over batch dim -> a1[b], a2[b]
// ---------------------------------------------------------------------------
__global__ void moe_softmax_batch(const float* __restrict__ s1s2,
                                  float* __restrict__ a1a2) {
  const int t = threadIdx.x;
  if (t >= 2 * NB) return;
  const int which = t >> 5, b = t & 31;
  const float* s = s1s2 + which * NB;
  float mx = -3.4e38f;
  for (int i = 0; i < NB; ++i) mx = fmaxf(mx, s[i]);
  float den = 0.f;
  for (int i = 0; i < NB; ++i) den += expf(s[i] - mx);
  a1a2[which * NB + b] = expf(s[b] - mx) / den;
}

// ---------------------------------------------------------------------------
// K8: final 3x3 conv over concat[moe, a1*moe, a2*sh] (concat folded into the
// LDS A-tile build). Implicit GEMM M=1024 N=128 K=3456. grid = (32, B).
// ---------------------------------------------------------------------------
__global__ void moe_final_conv_wmma(const float* __restrict__ moe,
                                    const float* __restrict__ sh,
                                    const __bf16* __restrict__ pwcc,
                                    const float* __restrict__ bcc,
                                    const float* __restrict__ a1a2,
                                    float* __restrict__ out) {
  __shared__ unsigned ldsU[CHUNK_CC * 512];
  const int tid = threadIdx.x;
  const int lane = tid & 31, w = tid >> 5;
  const int m0 = blockIdx.x * 32;
  const int b  = blockIdx.y;
  const float* mb = moe + (size_t)b * NC2 * NM;
  const float* sb = sh  + (size_t)b * NC2 * NM;
  const float a1 = a1a2[b], a2 = a1a2[NB + b];
  const __bf16* pw0 = pwcc + (size_t)(2 * w + 0) * KTI_CC * 512;
  const __bf16* pw1 = pwcc + (size_t)(2 * w + 1) * KTI_CC * 512;

  v8f acc00 = {}, acc01 = {}, acc10 = {}, acc11 = {};
  for (int kt0 = 0; kt0 < KTI_CC; kt0 += CHUNK_CC) {
    __builtin_prefetch(&pw0[(size_t)kt0 * 512 + lane * 16], 0, 0);
    __builtin_prefetch(&pw1[(size_t)kt0 * 512 + lane * 16], 0, 0);
    for (int j = tid; j < CHUNK_CC * 512; j += 128) {
      const int ktl = j >> 9;
      const int mi  = (j >> 8) & 1;
      const int rem = j & 255;
      const int li  = rem >> 3;
      const int h   = (rem & 7) << 1;
      const int m = m0 + mi * 16 + (li & 15);
      const int oy = m >> 5, ox = m & 31;
      const int k0 = (kt0 + ktl) * 32 + kA(li, h);
      float v[2];
#pragma unroll
      for (int q = 0; q < 2; ++q) {
        const int k  = k0 + q;                // k = cin*9 + di*3 + dj
        const int ci = k / 9;
        const int r  = k - ci * 9;
        const int di = r / 3, dj = r - (r / 3) * 3;
        const int grp = ci >> 7;              // 0: moe, 1: a1*moe, 2: a2*sh
        const int c   = ci & (NC2 - 1);
        const int iy = oy - 1 + di, ix = ox - 1 + dj;
        float av = 0.0f;
        if (iy >= 0 && iy < HM && ix >= 0 && ix < HM) {
          const int off = (c << 10) + (iy << 5) + ix;
          av = (grp == 2) ? sb[off] * a2 : mb[off] * ((grp == 1) ? a1 : 1.0f);
        }
        v[q] = av;
      }
      ldsU[j] = pack2bf(v[0], v[1]);
    }
    __syncthreads();
    for (int c = 0; c < CHUNK_CC; ++c) {
      const v16bf a0 = *(const v16bf*)&ldsU[((c * 2 + 0) * 32 + lane) * 8];
      const v16bf a1f = *(const v16bf*)&ldsU[((c * 2 + 1) * 32 + lane) * 8];
      const v16bf b0 = *(const v16bf*)&pw0[((size_t)(kt0 + c) * 32 + lane) * 16];
      const v16bf b1 = *(const v16bf*)&pw1[((size_t)(kt0 + c) * 32 + lane) * 16];
      acc00 = WMMA_BF16(a0,  b0, acc00);
      acc01 = WMMA_BF16(a0,  b1, acc01);
      acc10 = WMMA_BF16(a1f, b0, acc10);
      acc11 = WMMA_BF16(a1f, b1, acc11);
    }
    __syncthreads();
  }

  const int mlo = m0 + ((lane >> 4) << 3);
  const int nlo = (lane & 15);
  v8f accs[4] = {acc00, acc01, acc10, acc11};
#pragma unroll
  for (int t = 0; t < 4; ++t) {
    const int mi = t >> 1, nj = t & 1;
    const int n = (2 * w + nj) * 16 + nlo;
    const int mbase = mlo + mi * 16;
    const float bias = bcc[n];
    float* oo = out + ((size_t)b * NC2 + n) * NM;
#pragma unroll
    for (int r = 0; r < 8; ++r) oo[mbase + r] = accs[t][r] + bias;
  }
}

// ---------------------------------------------------------------------------
extern "C" void kernel_launch(void* const* d_in, const int* in_sizes, int n_in,
                              void* d_out, int out_size, void* d_ws, size_t ws_size,
                              hipStream_t stream) {
  (void)in_sizes; (void)n_in; (void)out_size; (void)ws_size;
  const float* x     = (const float*)d_in[0];
  const float* wgate = (const float*)d_in[1];
  const float* Wc    = (const float*)d_in[2];
  const float* bc    = (const float*)d_in[3];
  const float* gamma = (const float*)d_in[4];
  const float* beta  = (const float*)d_in[5];
  const float* Wsh   = (const float*)d_in[6];
  const float* bsh   = (const float*)d_in[7];
  const float* Wcc   = (const float*)d_in[8];
  const float* bcc   = (const float*)d_in[9];

  // workspace layout (float units; tile buffers 32B aligned)
  float* wsf    = (float*)d_ws;
  float* logits = wsf;                                    // 256
  int*   sel    = (int*)(wsf + 256);                      // 64
  float* gates  = wsf + 320;                              // 64
  float* s1s2   = wsf + 384;                              // 64
  float* a1a2   = wsf + 448;                              // 64
  float* yex    = wsf + 512;                              // 64*128*1024
  float* moe    = yex + (size_t)NB * NK * NC2 * NM;       // 32*128*1024
  float* shv    = moe + (size_t)NB * NC2 * NM;            // 32*128*1024
  __bf16* pwc   = (__bf16*)(shv + (size_t)NB * NC2 * NM); // 8*8*98*512 bf16
  __bf16* pwsh  = pwc  + (size_t)NE * 8 * KTI_EXP * 512;  // 8*32*512 bf16
  __bf16* pwcc  = pwsh + (size_t)8 * KTI_SH * 512;        // 8*108*512 bf16

  moe_pack_wc <<<(NE * 8 * KTI_EXP * 512 + 255) / 256, 256, 0, stream>>>(Wc, pwc);
  moe_pack_wsh<<<(8 * KTI_SH * 512 + 255) / 256, 256, 0, stream>>>(Wsh, pwsh);
  moe_pack_wcc<<<(8 * KTI_CC * 512 + 255) / 256, 256, 0, stream>>>(Wcc, pwcc);

  moe_gating_logits<<<NB, 256, 0, stream>>>(x, wgate, logits);
  moe_top2<<<1, 32, 0, stream>>>(logits, sel, gates);
  moe_expert_conv_wmma<<<dim3(NM / 32, NB * NK), 128, 0, stream>>>(
      x, pwc, bc, sel, yex);
  moe_ln_combine<<<(NB * NM + 255) / 256, 256, 0, stream>>>(
      yex, sel, gates, gamma, beta, moe);
  moe_shared_expert_wmma<<<dim3(NM / 32, NB), 128, 0, stream>>>(
      x, pwsh, bsh, shv);
  moe_batch_dots<<<NB, 256, 0, stream>>>(moe, shv, s1s2);
  moe_softmax_batch<<<1, 64, 0, stream>>>(s1s2, a1a2);
  moe_final_conv_wmma<<<dim3(NM / 32, NB), 128, 0, stream>>>(
      moe, shv, pwcc, bcc, a1a2, (float*)d_out);
}
